// GraphNet_74053826118282
// MI455X (gfx1250) — matrix operand order repaired and checked by reading, not compiled, not run on hardware
//
#include <hip/hip_runtime.h>
#include <hip/hip_bf16.h>

typedef __attribute__((ext_vector_type(16))) _Float16 v16h;
typedef __attribute__((ext_vector_type(8)))  _Float16 v8h;
typedef __attribute__((ext_vector_type(8)))  float    v8f;
typedef unsigned int u32;
typedef __attribute__((ext_vector_type(4))) unsigned int v4u;
typedef __attribute__((ext_vector_type(8))) int         v8i_t;
typedef __attribute__((ext_vector_type(4))) int         v4i_t;

#define CDIV(a,b) (((a)+(b)-1)/(b))

constexpr int NN     = 10000;    // nodes
constexpr int NE     = 160000;   // edges
constexpr int KP     = 5000;     // kept nodes (alpha = 0.5)
constexpr int SORT_N = 16384;    // next pow2 >= NN

// flat output layout (floats): x1 | new_ei | perm | fitness | loss1
constexpr int OFF_X1   = 0;
constexpr int OFF_EI   = KP;                 // 5000
constexpr int OFF_PERM = OFF_EI + 2 * NE;    // 325000
constexpr int OFF_FIT  = OFF_PERM + KP;      // 330000
constexpr int OFF_LOSS = OFF_FIT + NN;       // 340000

// ---------------- utility kernels ----------------
__global__ void k_fill_f32(float* p, float v, int n) {
  int i = blockIdx.x * blockDim.x + threadIdx.x;
  if (i < n) p[i] = v;
}
__global__ void k_fill_i32(int* p, int v, int n) {
  int i = blockIdx.x * blockDim.x + threadIdx.x;
  if (i < n) p[i] = v;
}
__global__ void k_cvt_f16(const float* __restrict__ x, _Float16* __restrict__ y, int n) {
  int i = blockIdx.x * blockDim.x + threadIdx.x;
  if (i < n) y[i] = (_Float16)x[i];
}
// W is [K,N] row-major fp32; Wt is [N,K] row-major f16
__global__ void k_cvt_wt(const float* __restrict__ W, _Float16* __restrict__ Wt, int K, int N) {
  int i = blockIdx.x * blockDim.x + threadIdx.x;
  if (i < K * N) {
    int n = i / K, k = i % K;
    Wt[i] = (_Float16)W[k * N + n];
  }
}
__global__ void k_rsqrt(const float* __restrict__ a, float* __restrict__ b, int n) {
  int i = blockIdx.x * blockDim.x + threadIdx.x;
  if (i < n) b[i] = rsqrtf(a[i]);
}

// ---------------- Tensor Data Mover helper ----------------
// 2-D tile load: tile_rows x k_elems f16 elements, row stride k_elems,
// zero-fill for rows beyond rows_remaining (OOB handled by the TDM).
// Descriptor packing per CDNA5 ISA section 8.3/8.4 (groups 2/3 unused -> zero).
// This toolchain exposes the 6-arg builtin: (u32x4, i32x8, i32x4, i32x4, i32x8, cpol).
__device__ inline void tdm_load_2d(u32 lds_off, const _Float16* gptr,
                                   u32 k_elems, u32 rows_remaining, u32 tile_rows) {
  unsigned long long ga = (unsigned long long)(size_t)gptr;
  v4u g0;
  g0[0] = 1u;                                           // count=1, user descriptor
  g0[1] = lds_off;                                      // LDS byte address
  g0[2] = (u32)ga;                                      // global addr [31:0]
  g0[3] = (u32)((ga >> 32) & 0x1FFFFFFu) | (2u << 30);  // addr[56:32] | type=2
  v8i_t g1;
  g1[0] = (int)(1u << 16);                              // data_size=1 (2 bytes/elem)
  g1[1] = (int)((k_elems & 0xFFFFu) << 16);             // tensor_dim0 lo16
  g1[2] = (int)(((k_elems >> 16) & 0xFFFFu) |           // tensor_dim0 hi16
                ((rows_remaining & 0xFFFFu) << 16));    // tensor_dim1 lo16
  g1[3] = (int)(((rows_remaining >> 16) & 0xFFFFu) |    // tensor_dim1 hi16
                ((k_elems & 0xFFFFu) << 16));           // tile_dim0 = K
  g1[4] = (int)(tile_rows & 0xFFFFu);                   // tile_dim1 (tile_dim2=0)
  g1[5] = (int)k_elems;                                 // tensor_dim0_stride lo32
  g1[6] = 0;                                            // stride0 hi | stride1 lo
  g1[7] = 0;
  v4i_t z4 = {};
  v8i_t z8 = {};
  __builtin_amdgcn_tensor_load_to_lds(g0, g1, z4, z4, z8, 0);
}

// ---------------- WMMA GEMM: C[M,N] = A[M,K] * Bt[N,K]^T ----------------
// 128-thread block (4 waves) computes a 64x64 C tile. Each wave TDM-loads one
// 16-row quarter of the A tile and of the B tile into LDS, then all waves feed
// v_wmma_f32_16x16x32_f16 from LDS. A fragment reused across 4 B fragments.
__device__ inline v16h frag_pair(const _Float16* p0, const _Float16* p1) {
  v8h lo = *(const v8h*)p0;
  v8h hi = *(const v8h*)p1;
  v16h r;
#pragma unroll
  for (int i = 0; i < 8; ++i) { r[i] = lo[i]; r[i + 8] = hi[i]; }
  return r;
}
// A 16x32 f16 fragment: lane m=l&15; halves 0..7 at K=k0+8*sel, halves 8..15 at +16
__device__ inline v16h frag_a(const _Float16* A, int ld, int row, int k0, int lane) {
  int sel = lane >> 4;
  const _Float16* p = A + (size_t)row * ld + k0 + sel * 8;
  return frag_pair(p, p + 16);
}
// B 32x16 f16 fragment from Bt rows: lane n=l&15; halves 0..7 at K=k0+8*sel, halves 8..15 at +16
__device__ inline v16h frag_b(const _Float16* Bt, int ld, int col, int k0, int lane) {
  int sel = lane >> 4;
  const _Float16* p = Bt + (size_t)col * ld + k0 + sel * 8;
  return frag_pair(p, p + 16);
}

__global__ void __launch_bounds__(128)
k_gemm_f16(const _Float16* __restrict__ A, const _Float16* __restrict__ Bt,
           float* __restrict__ C, int M, int K, int N) {
  __shared__ _Float16 shA[64 * 256];
  __shared__ _Float16 shB[64 * 256];
  const int lane  = threadIdx.x & 31;
  const int wv    = threadIdx.x >> 5;       // wave id 0..3
  const int tileM = blockIdx.x * 64;
  const int tileN = blockIdx.y * 64;

  // async tile staging: each wave DMAs disjoint 16-row quarters
  {
    const int rowA = tileM + 16 * wv;
    const u32 remA = (rowA < M) ? (u32)(M - rowA) : 0u;
    tdm_load_2d((u32)(size_t)&shA[16 * wv * K], A + (size_t)rowA * K, (u32)K, remA, 16u);
    const int rowB = tileN + 16 * wv;
    tdm_load_2d((u32)(size_t)&shB[16 * wv * K], Bt + (size_t)rowB * K, (u32)K, (u32)(N - rowB), 16u);
  }
  __builtin_amdgcn_s_wait_tensorcnt(0);
  __syncthreads();

  v8f acc0 = {}, acc1 = {}, acc2 = {}, acc3 = {};
  const int ar = 16 * wv + (lane & 15);
  for (int k0 = 0; k0 < K; k0 += 32) {
    v16h a  = frag_a(shA, K, ar, k0, lane);
    v16h b0 = frag_b(shB, K,  0 + (lane & 15), k0, lane);
    v16h b1 = frag_b(shB, K, 16 + (lane & 15), k0, lane);
    v16h b2 = frag_b(shB, K, 32 + (lane & 15), k0, lane);
    v16h b3 = frag_b(shB, K, 48 + (lane & 15), k0, lane);
    acc0 = __builtin_amdgcn_wmma_f32_16x16x32_f16(false, a, false, b0, (short)0, acc0, false, false);
    acc1 = __builtin_amdgcn_wmma_f32_16x16x32_f16(false, a, false, b1, (short)0, acc1, false, false);
    acc2 = __builtin_amdgcn_wmma_f32_16x16x32_f16(false, a, false, b2, (short)0, acc2, false, false);
    acc3 = __builtin_amdgcn_wmma_f32_16x16x32_f16(false, a, false, b3, (short)0, acc3, false, false);
  }
  // C layout: lane n=l&15, VGPR r holds M = r + 8*sel
  const int n0  = tileN + (lane & 15);
  const int sel = lane >> 4;
#pragma unroll
  for (int r = 0; r < 8; ++r) {
    const int row = tileM + 16 * wv + r + 8 * sel;
    if (row < M) {
      const size_t ro = (size_t)row * N;
      C[ro + n0 +  0] = acc0[r];
      C[ro + n0 + 16] = acc1[r];
      C[ro + n0 + 32] = acc2[r];
      C[ro + n0 + 48] = acc3[r];
    }
  }
}

// ---------------- graph aggregation ----------------
__global__ void k_deg_count(float* deg, const int* __restrict__ dst, int ne) {
  int e = blockIdx.x * blockDim.x + threadIdx.x;
  if (e < ne) atomicAdd(&deg[dst[e]], 1.0f);
}
// gridDim.x = NE, blockDim.x = C
__global__ void k_edge_agg(const float* __restrict__ H, float* __restrict__ AGG,
                           const int* __restrict__ src, const int* __restrict__ dst,
                           const float* __restrict__ dinv, int C) {
  int e = blockIdx.x;
  int c = threadIdx.x;
  int s = src[e], d = dst[e];
  float nrm = dinv[s] * dinv[d];
  atomicAdd(&AGG[(size_t)d * C + c], H[(size_t)s * C + c] * nrm);
}
__global__ void k_finalize(const float* __restrict__ AGG, const float* __restrict__ H,
                           const float* __restrict__ dinv, const float* __restrict__ b,
                           float* __restrict__ OUT, int M, int C) {
  int i = blockIdx.x * blockDim.x + threadIdx.x;
  if (i >= M * C) return;
  int node = i / C, c = i % C;
  float di = dinv[node];
  float v = AGG[i] + H[i] * di * di + b[c];
  OUT[i] = v > 0.f ? v : 0.f;
}

// ---------------- pooling ----------------
// one wave per node: score = h3[node,:] . p ; fitness = sigmoid
__global__ void k_score(const float* __restrict__ H3, const float* __restrict__ p,
                        float* __restrict__ sortK, int* __restrict__ sortV,
                        float* __restrict__ fit_out, int n, int C) {
  int wid  = (blockIdx.x * blockDim.x + threadIdx.x) >> 5;
  int lane = threadIdx.x & 31;
  if (wid >= n) return;
  const float* row = H3 + (size_t)wid * C;
  float s = 0.f;
  for (int j = lane; j < C; j += 32) s += row[j] * p[j];
  for (int o = 16; o; o >>= 1) s += __shfl_xor(s, o, 32);
  if (lane == 0) {
    float f = 1.f / (1.f + expf(-s));
    sortK[wid] = f; sortV[wid] = wid; fit_out[wid] = f;
  }
}
__global__ void k_sort_pad(float* sortK, int* sortV) {
  int i = blockIdx.x * blockDim.x + threadIdx.x + NN;
  if (i < SORT_N) { sortK[i] = -1.0f; sortV[i] = i; }
}
// global bitonic, descending
__global__ void k_bitonic(float* key, int* val, int kk, int j) {
  int i = blockIdx.x * blockDim.x + threadIdx.x;
  int ixj = i ^ j;
  if (ixj > i) {
    bool up = ((i & kk) == 0);
    float ki = key[i], kj = key[ixj];
    bool sw = up ? (ki < kj) : (ki > kj);
    if (sw) {
      key[i] = kj; key[ixj] = ki;
      int t = val[i]; val[i] = val[ixj]; val[ixj] = t;
    }
  }
}
__global__ void k_nmap_set(const int* __restrict__ sortV, int* __restrict__ nmap) {
  int j = blockIdx.x * blockDim.x + threadIdx.x;
  if (j < KP) nmap[sortV[j]] = j;
}
// gridDim = KP, blockDim = 128
__global__ void k_xpool(const float* __restrict__ H3, const float* __restrict__ sortK,
                        const int* __restrict__ sortV, float* __restrict__ xp) {
  int j = blockIdx.x, c = threadIdx.x;
  xp[(size_t)j * 128 + c] = H3[(size_t)sortV[j] * 128 + c] * sortK[j];
}
__global__ void k_perm_out(const int* __restrict__ sortV, float* __restrict__ out_perm) {
  int j = blockIdx.x * blockDim.x + threadIdx.x;
  if (j < KP) out_perm[j] = (float)sortV[j];
}
__global__ void k_edge_remap(const int* __restrict__ src, const int* __restrict__ dst,
                             const int* __restrict__ nmap, int* __restrict__ nsrc,
                             int* __restrict__ ndst, float* __restrict__ ew,
                             float* __restrict__ out_ei) {
  int e = blockIdx.x * blockDim.x + threadIdx.x;
  if (e >= NE) return;
  int s = nmap[src[e]], d = nmap[dst[e]];
  bool keep = (s >= 0) && (d >= 0);
  s = keep ? s : 0; d = keep ? d : 0;
  nsrc[e] = s; ndst[e] = d; ew[e] = keep ? 1.f : 0.f;
  out_ei[e] = (float)s; out_ei[NE + e] = (float)d;
}
__global__ void k_loss(const float* __restrict__ sortK, float* acc, int k) {
  __shared__ float sm[256];
  int i = blockIdx.x * 256 + threadIdx.x;
  sm[threadIdx.x] = (i < k) ? logf(sortK[i] + 1e-8f) : 0.f;
  __syncthreads();
  for (int o = 128; o; o >>= 1) {
    if ((int)threadIdx.x < o) sm[threadIdx.x] += sm[threadIdx.x + o];
    __syncthreads();
  }
  if (threadIdx.x == 0) atomicAdd(acc, sm[0]);
}
__global__ void k_loss_final(const float* acc, float* out) { out[0] = -acc[0] / (float)KP; }

// ---------------- final scalar GCN layer ----------------
__global__ void k_h4(const float* __restrict__ xp, const float* __restrict__ W4,
                     float* __restrict__ h4, int n) {
  int wid  = (blockIdx.x * blockDim.x + threadIdx.x) >> 5;
  int lane = threadIdx.x & 31;
  if (wid >= n) return;
  const float* row = xp + (size_t)wid * 128;
  float s = 0.f;
  for (int j = lane; j < 128; j += 32) s += row[j] * W4[j];
  for (int o = 16; o; o >>= 1) s += __shfl_xor(s, o, 32);
  if (lane == 0) h4[wid] = s;
}
__global__ void k_deg2_count(float* deg2, const int* __restrict__ ndst,
                             const float* __restrict__ ew, int ne) {
  int e = blockIdx.x * blockDim.x + threadIdx.x;
  if (e < ne) atomicAdd(&deg2[ndst[e]], ew[e]);
}
__global__ void k_edge_agg2(const float* __restrict__ h4, float* __restrict__ agg2,
                            const int* __restrict__ nsrc, const int* __restrict__ ndst,
                            const float* __restrict__ dinv2, const float* __restrict__ ew, int ne) {
  int e = blockIdx.x * blockDim.x + threadIdx.x;
  if (e >= ne) return;
  int s = nsrc[e], d = ndst[e];
  atomicAdd(&agg2[d], h4[s] * dinv2[s] * dinv2[d] * ew[e]);
}
__global__ void k_final2(const float* __restrict__ agg2, const float* __restrict__ h4,
                         const float* __restrict__ dinv2, const float* __restrict__ b4,
                         float* __restrict__ out_x1, int n) {
  int i = blockIdx.x * blockDim.x + threadIdx.x;
  if (i >= n) return;
  float di = dinv2[i];
  float v = agg2[i] + h4[i] * di * di + b4[0];
  v = v > 0.f ? v : 0.f;
  out_x1[i] = 1.f / (1.f + expf(-v));
}

// ---------------- host ----------------
static inline size_t align256(size_t x) { return (x + 255) & ~(size_t)255; }

extern "C" void kernel_launch(void* const* d_in, const int* in_sizes, int n_in,
                              void* d_out, int out_size, void* d_ws, size_t ws_size,
                              hipStream_t stream) {
  (void)in_sizes; (void)n_in; (void)out_size; (void)ws_size;
  const float* x  = (const float*)d_in[0];
  const int*   ei = (const int*)d_in[1];
  const float* W1 = (const float*)d_in[2];  const float* b1 = (const float*)d_in[3];
  const float* W2 = (const float*)d_in[4];  const float* b2 = (const float*)d_in[5];
  const float* W3 = (const float*)d_in[6];  const float* b3 = (const float*)d_in[7];
  const float* pS = (const float*)d_in[8];
  const float* W4 = (const float*)d_in[9];  const float* b4 = (const float*)d_in[10];
  const int* src = ei;
  const int* dst = ei + NE;
  float* out = (float*)d_out;

  // workspace carve-out
  char* w = (char*)d_ws;
  auto alloc = [&](size_t bytes) { char* p = w; w += align256(bytes); return p; };
  _Float16* A16 = (_Float16*)alloc((size_t)NN * 256 * 2);
  _Float16* Wt  = (_Float16*)alloc((size_t)256 * 256 * 2);
  float* H    = (float*)alloc((size_t)NN * 256 * 4);
  float* AGG  = (float*)alloc((size_t)NN * 256 * 4);
  float* O1   = (float*)alloc((size_t)NN * 256 * 4);
  float* O2   = (float*)alloc((size_t)NN * 256 * 4);
  float* deg  = (float*)alloc(NN * 4);
  float* dinv = (float*)alloc(NN * 4);
  float* sortK = (float*)alloc(SORT_N * 4);
  int*   sortV = (int*)alloc(SORT_N * 4);
  int*   nmap  = (int*)alloc(NN * 4);
  float* xp    = (float*)alloc((size_t)KP * 128 * 4);
  int*   nsrc  = (int*)alloc(NE * 4);
  int*   ndst  = (int*)alloc(NE * 4);
  float* ew    = (float*)alloc(NE * 4);
  float* h4    = (float*)alloc(KP * 4);
  float* agg2  = (float*)alloc(KP * 4);
  float* deg2  = (float*)alloc(KP * 4);
  float* dinv2 = (float*)alloc(KP * 4);
  float* lossA = (float*)alloc(4);

  // degrees (shared by layers 1-3; edge_w == 1)
  k_fill_f32<<<CDIV(NN, 256), 256, 0, stream>>>(deg, 1.0f, NN);
  k_deg_count<<<CDIV(NE, 256), 256, 0, stream>>>(deg, dst, NE);
  k_rsqrt<<<CDIV(NN, 256), 256, 0, stream>>>(deg, dinv, NN);

  auto run_layer = [&](const float* X, const float* Wf, const float* bf,
                       float* OUT, int Cin, int Cout) {
    k_cvt_f16<<<CDIV(NN * Cin, 256), 256, 0, stream>>>(X, A16, NN * Cin);
    k_cvt_wt<<<CDIV(Cin * Cout, 256), 256, 0, stream>>>(Wf, Wt, Cin, Cout);
    k_gemm_f16<<<dim3(CDIV(NN, 64), Cout / 64), 128, 0, stream>>>(A16, Wt, H, NN, Cin, Cout);
    k_fill_f32<<<CDIV(NN * Cout, 256), 256, 0, stream>>>(AGG, 0.f, NN * Cout);
    k_edge_agg<<<NE, Cout, 0, stream>>>(H, AGG, src, dst, dinv, Cout);
    k_finalize<<<CDIV(NN * Cout, 256), 256, 0, stream>>>(AGG, H, dinv, bf, OUT, NN, Cout);
  };
  run_layer(x,  W1, b1, O1, 128, 256);
  run_layer(O1, W2, b2, O2, 256, 256);
  run_layer(O2, W3, b3, O1, 256, 128);   // h3 lives in O1 [NN,128]

  // pooling: score + sort + remap
  k_score<<<CDIV(NN, 8), 256, 0, stream>>>(O1, pS, sortK, sortV, out + OFF_FIT, NN, 128);
  k_sort_pad<<<CDIV(SORT_N - NN, 256), 256, 0, stream>>>(sortK, sortV);
  for (int kk = 2; kk <= SORT_N; kk <<= 1)
    for (int j = kk >> 1; j > 0; j >>= 1)
      k_bitonic<<<SORT_N / 256, 256, 0, stream>>>(sortK, sortV, kk, j);
  k_fill_i32<<<CDIV(NN, 256), 256, 0, stream>>>(nmap, -1, NN);
  k_nmap_set<<<CDIV(KP, 256), 256, 0, stream>>>(sortV, nmap);
  k_xpool<<<KP, 128, 0, stream>>>(O1, sortK, sortV, xp);
  k_perm_out<<<CDIV(KP, 256), 256, 0, stream>>>(sortV, out + OFF_PERM);
  k_edge_remap<<<CDIV(NE, 256), 256, 0, stream>>>(src, dst, nmap, nsrc, ndst, ew, out + OFF_EI);
  k_fill_f32<<<1, 32, 0, stream>>>(lossA, 0.f, 1);
  k_loss<<<CDIV(KP, 256), 256, 0, stream>>>(sortK, lossA, KP);
  k_loss_final<<<1, 32, 0, stream>>>(lossA, out + OFF_LOSS);

  // final scalar GCN layer on pooled graph
  k_h4<<<CDIV(KP, 8), 256, 0, stream>>>(xp, W4, h4, KP);
  k_fill_f32<<<CDIV(KP, 256), 256, 0, stream>>>(deg2, 1.0f, KP);
  k_deg2_count<<<CDIV(NE, 256), 256, 0, stream>>>(deg2, ndst, ew, NE);
  k_rsqrt<<<CDIV(KP, 256), 256, 0, stream>>>(deg2, dinv2, KP);
  k_fill_f32<<<CDIV(KP, 256), 256, 0, stream>>>(agg2, 0.f, KP);
  k_edge_agg2<<<CDIV(NE, 256), 256, 0, stream>>>(h4, agg2, nsrc, ndst, dinv2, ew, NE);
  k_final2<<<CDIV(KP, 256), 256, 0, stream>>>(agg2, h4, dinv2, b4, out + OFF_X1, KP);
}
